// DkNNModel_11888469476367
// MI455X (gfx1250) — matrix operand order repaired and checked by reading, not compiled, not run on hardware
//
#include <hip/hip_runtime.h>
#include <stdint.h>

// Problem constants (match reference)
#define LL 4
#define BB 1024
#define NN 50000
#define DD 256
#define KK 75
#define CC 10
#define NB_CALI 750

// Tiling
#define WAVES 4
#define QW 16                 // queries per wave (WMMA M)
#define QB (QW * WAVES)       // 64 queries per block
#define NT 32                 // train points per LDS tile
#define NSLICE 4              // N split across blocks for occupancy
#define SL ((NN + NSLICE - 1) / NSLICE)   // 12500

// Dynamic LDS layout (bytes)
#define TILE_BYTES (NT * DD * 2)                      // 16384 per buffer
#define OFF_TLO    (2 * TILE_BYTES)                   // 32768
#define OFF_T2     (4 * TILE_BYTES)                   // 65536
#define OFF_LAB    (OFF_T2 + 2 * NT * 4)              // 65792
#define OFF_KEYS   (OFF_LAB + 2 * NT * 4)             // 66048
#define OFF_LIST   (OFF_KEYS + QB * NT * 4)           // 74240
#define SMEM_BYTES (OFF_LIST + QB * KK * 4)           // 93440

typedef _Float16 half_t;
typedef __attribute__((ext_vector_type(16))) _Float16 v16h;
typedef __attribute__((ext_vector_type(8)))  _Float16 v8h;
typedef __attribute__((ext_vector_type(8)))  float    v8f;

union V16 {
    v16h v;
    v8h  h[2];
};

// ---------------------------------------------------------------------------
// CDNA5 async global->LDS copies (ASYNCcnt tracked, no VGPR staging).
// Generic LDS pointer low 32 bits are the LDS byte offset (ISA 10.2:
// LDS address is the low 32 bits), which is what VDST of the async op expects.
// ---------------------------------------------------------------------------
static __device__ __forceinline__ void async_cp_b128(void* lds, const void* g) {
    unsigned loff = (unsigned)(uintptr_t)lds;
    unsigned long long ga = (unsigned long long)(uintptr_t)g;
    asm volatile("global_load_async_to_lds_b128 %0, %1, off"
                 :: "v"(loff), "v"(ga) : "memory");
}
static __device__ __forceinline__ void async_cp_b32(void* lds, const void* g) {
    unsigned loff = (unsigned)(uintptr_t)lds;
    unsigned long long ga = (unsigned long long)(uintptr_t)g;
    asm volatile("global_load_async_to_lds_b32 %0, %1, off"
                 :: "v"(loff), "v"(ga) : "memory");
}
static __device__ __forceinline__ void wait_async0() {
    asm volatile("s_wait_asynccnt 0x0" ::: "memory");
}

// ---------------------------------------------------------------------------
// Prep: per-row L2 normalize, subtract per-layer center, split f32 -> f16 hi+lo,
// optionally emit sum((x/||x|| - c)^2) per row. One wave32 per row of 256.
// ---------------------------------------------------------------------------
__global__ __launch_bounds__(256)
void prep_rows(const float* __restrict__ src, const float* __restrict__ centers,
               half_t* __restrict__ hi, half_t* __restrict__ lo,
               float* __restrict__ sq, int rowsPerLayer, int totalRows) {
    const int warp = threadIdx.x >> 5;
    const int lane = threadIdx.x & 31;
    const long row = (long)blockIdx.x * (blockDim.x >> 5) + warp;
    if (row >= totalRows) return;
    const int l = (int)(row / rowsPerLayer);
    const float* x = src + row * (long)DD;
    const float* c = centers + (long)l * DD;

    float v[8];
    float ss = 0.f;
#pragma unroll
    for (int i = 0; i < 8; ++i) {
        v[i] = x[lane * 8 + i];
        ss += v[i] * v[i];
    }
#pragma unroll
    for (int m = 16; m >= 1; m >>= 1) ss += __shfl_xor(ss, m, 32);

    float inv = rsqrtf(ss);
    inv = inv * (1.5f - 0.5f * ss * inv * inv);   // Newton step -> ~f32 accurate

    float t2 = 0.f;
    v8h hv, lv;
#pragma unroll
    for (int i = 0; i < 8; ++i) {
        float w = v[i] * inv - c[lane * 8 + i];
        t2 += w * w;
        half_t h = (half_t)w;
        hv[i] = h;
        lv[i] = (half_t)(w - (float)h);
    }
#pragma unroll
    for (int m = 16; m >= 1; m >>= 1) t2 += __shfl_xor(t2, m, 32);

    *(v8h*)(hi + row * (long)DD + lane * 8) = hv;
    *(v8h*)(lo + row * (long)DD + lane * 8) = lv;
    if (sq != nullptr && lane == 0) sq[row] = t2;
}

// ---------------------------------------------------------------------------
// Issue async copies for one NT-row tile into an LDS buffer (zero-pad tail).
// ---------------------------------------------------------------------------
static __device__ __forceinline__ void issue_tile(
    half_t* dsthi, half_t* dstlo, float* dst2, int* dslab,
    const half_t* __restrict__ thi, const half_t* __restrict__ tlo,
    const float* __restrict__ t2g, const int* __restrict__ labels,
    long tRowBase, int tbase, int nend, int tid) {
    for (int vi = tid; vi < NT * DD / 8; vi += 128) {
        const int r    = vi >> 5;                 // 32 x 16B vectors per row
        const int cidx = (vi & 31) * 8;
        const int n    = tbase + r;
        if (n < nend) {
            async_cp_b128(&dsthi[r * DD + cidx], thi + (tRowBase + n) * (long)DD + cidx);
            async_cp_b128(&dstlo[r * DD + cidx], tlo + (tRowBase + n) * (long)DD + cidx);
        } else {
            const uint4 z = make_uint4(0, 0, 0, 0);
            *(uint4*)&dsthi[r * DD + cidx] = z;
            *(uint4*)&dstlo[r * DD + cidx] = z;
        }
    }
    if (tid < NT) {
        const int n = tbase + tid;
        if (n < nend) {
            async_cp_b32(&dst2[tid], t2g + tRowBase + n);
            async_cp_b32(&dslab[tid], labels + n);
        } else {
            dst2[tid]  = 0.f;
            dslab[tid] = 0;
        }
    }
}

// ---------------------------------------------------------------------------
// Fused split-f16 WMMA GEMM + per-slice top-75 selection.
// grid = (BB/QB, LL, NSLICE), block = 128 threads (4 wave32).
// Double-buffered LDS tiles filled by async-to-LDS; 4 independent WMMA chains.
// ---------------------------------------------------------------------------
__global__ __launch_bounds__(128)
void knn_kernel(const half_t* __restrict__ qhi, const half_t* __restrict__ qlo,
                const half_t* __restrict__ thi, const half_t* __restrict__ tlo,
                const float* __restrict__ t2g, const int* __restrict__ labels,
                unsigned* __restrict__ partial /* [L*B][NSLICE][KK] */) {
    extern __shared__ char smem[];
    half_t*   s_thi  = (half_t*)smem;                       // [2][NT*DD]
    half_t*   s_tlo  = (half_t*)(smem + OFF_TLO);           // [2][NT*DD]
    float*    s_t2   = (float*)(smem + OFF_T2);             // [2][NT]
    int*      s_lab  = (int*)(smem + OFF_LAB);              // [2][NT]
    unsigned* s_keys = (unsigned*)(smem + OFF_KEYS);        // [QB][NT]
    unsigned* s_list = (unsigned*)(smem + OFF_LIST);        // [QB][KK]

    const int tid   = threadIdx.x;
    const int lane  = tid & 31;
    const int w     = tid >> 5;
    const int l     = blockIdx.y;
    const int slice = blockIdx.z;
    const int qbase = blockIdx.x * QB;
    const int nbase = slice * SL;
    const int nend  = (nbase + SL < NN) ? (nbase + SL) : NN;

    // init per-query top-K lists (key 0 < any real score's monotone encoding)
    if (tid < QB) {
#pragma unroll 1
        for (int i = 0; i < KK; ++i) s_list[tid * KK + i] = 0u;
    }
    unsigned curMin = 0u;

    // --- A fragments resident in VGPRs for the whole slice loop -------------
    // 16-bit A 16x32 layout: row M = lane&15; lanes<16 hold K {0..7,16..23},
    // lanes>=16 hold K {8..15,24..31} => two contiguous 16B chunks per step.
    const int  mrow  = lane & 15;
    const int  khalf = (lane >> 4) ? 8 : 0;
    const long qrow  = ((long)l * BB + qbase + w * QW + mrow) * DD;
    V16 a_hi[8], a_lo[8];
#pragma unroll
    for (int ks = 0; ks < 8; ++ks) {
        const int kb = ks * 32;
        a_hi[ks].h[0] = *(const v8h*)(qhi + qrow + kb + khalf);
        a_hi[ks].h[1] = *(const v8h*)(qhi + qrow + kb + khalf + 16);
        a_lo[ks].h[0] = *(const v8h*)(qlo + qrow + kb + khalf);
        a_lo[ks].h[1] = *(const v8h*)(qlo + qrow + kb + khalf + 16);
    }

    const long tRowBase = (long)l * NN;

    // prologue: fill buffer 0
    int buf = 0;
    issue_tile(s_thi, s_tlo, s_t2, s_lab,
               thi, tlo, t2g, labels, tRowBase, nbase, nend, tid);
    wait_async0();
    __syncthreads();

    for (int tbase = nbase; tbase < nend; tbase += NT) {
        // prefetch next tile into the other buffer (async, overlaps WMMAs)
        const int nxt = tbase + NT;
        const int nb  = buf ^ 1;
        if (nxt < nend) {
            issue_tile(s_thi + nb * (NT * DD), s_tlo + nb * (NT * DD),
                       s_t2 + nb * NT, s_lab + nb * NT,
                       thi, tlo, t2g, labels, tRowBase, nxt, nend, tid);
        }

        // --- compute: 16 queries x 32 cols, 4 independent WMMA chains -------
        half_t* cthi = s_thi + buf * (NT * DD);
        half_t* ctlo = s_tlo + buf * (NT * DD);
        const int koff  = (lane >> 4) ? 8 : 0;      // B 32x16: lanes>=16 hold K {8..15,24..31}
        const int nloc0 = lane & 15;
        const int nloc1 = 16 + nloc0;
        v8f acc00 = {}, acc01 = {}, acc10 = {}, acc11 = {};
#pragma unroll
        for (int ks = 0; ks < 8; ++ks) {
            const int kb = ks * 32;
            V16 bh0, bl0, bh1, bl1;
            bh0.h[0] = *(const v8h*)&cthi[nloc0 * DD + kb + koff];
            bh0.h[1] = *(const v8h*)&cthi[nloc0 * DD + kb + koff + 16];
            bh1.h[0] = *(const v8h*)&cthi[nloc1 * DD + kb + koff];
            bh1.h[1] = *(const v8h*)&cthi[nloc1 * DD + kb + koff + 16];
            bl0.h[0] = *(const v8h*)&ctlo[nloc0 * DD + kb + koff];
            bl0.h[1] = *(const v8h*)&ctlo[nloc0 * DD + kb + koff + 16];
            bl1.h[0] = *(const v8h*)&ctlo[nloc1 * DD + kb + koff];
            bl1.h[1] = *(const v8h*)&ctlo[nloc1 * DD + kb + koff + 16];
            // split-f16 ~f32 GEMM; hi*hi and cross terms on separate chains
            acc00 = __builtin_amdgcn_wmma_f32_16x16x32_f16(
                false, a_hi[ks].v, false, bh0.v, (short)0, acc00, false, false);
            acc10 = __builtin_amdgcn_wmma_f32_16x16x32_f16(
                false, a_hi[ks].v, false, bh1.v, (short)0, acc10, false, false);
            acc01 = __builtin_amdgcn_wmma_f32_16x16x32_f16(
                false, a_hi[ks].v, false, bl0.v, (short)0, acc01, false, false);
            acc11 = __builtin_amdgcn_wmma_f32_16x16x32_f16(
                false, a_hi[ks].v, false, bl1.v, (short)0, acc11, false, false);
            acc01 = __builtin_amdgcn_wmma_f32_16x16x32_f16(
                false, a_lo[ks].v, false, bh0.v, (short)0, acc01, false, false);
            acc11 = __builtin_amdgcn_wmma_f32_16x16x32_f16(
                false, a_lo[ks].v, false, bh1.v, (short)0, acc11, false, false);
        }

        // scores -> monotone u32 keys with label packed in low 4 bits
        {
            const float t2v0 = s_t2[buf * NT + nloc0];
            const float t2v1 = s_t2[buf * NT + nloc1];
            const int   lab0 = s_lab[buf * NT + nloc0];
            const int   lab1 = s_lab[buf * NT + nloc1];
            const int   n0   = tbase + nloc0;
            const int   n1   = tbase + nloc1;
            const int   mofs = (lane >> 4) ? 8 : 0;
#pragma unroll
            for (int r = 0; r < 8; ++r) {
                const int m = r + mofs;              // C/D layout row
                float sc0 = 2.f * (acc00[r] + acc01[r]) - t2v0;  // neg_d2 (+const)
                float sc1 = 2.f * (acc10[r] + acc11[r]) - t2v1;
                unsigned u0 = __float_as_uint(sc0);
                unsigned u1 = __float_as_uint(sc1);
                unsigned m0 = (u0 & 0x80000000u) ? ~u0 : (u0 | 0x80000000u);
                unsigned m1 = (u1 & 0x80000000u) ? ~u1 : (u1 | 0x80000000u);
                s_keys[(w * QW + m) * NT + nloc0] =
                    (n0 < nend) ? ((m0 & ~15u) | (unsigned)lab0) : 0u;
                s_keys[(w * QW + m) * NT + nloc1] =
                    (n1 < nend) ? ((m1 & ~15u) | (unsigned)lab1) : 0u;
            }
        }
        __syncthreads();   // keys visible to owners

        // --- selection: one owner thread per query, cached-min threshold ----
        if (tid < QB) {
            unsigned* list = &s_list[tid * KK];
#pragma unroll 1
            for (int j = 0; j < NT; ++j) {
                const unsigned kk = s_keys[tid * NT + j];
                if (kk > curMin) {                 // rare after warm-up
                    int mi = 0; unsigned mv = list[0];
#pragma unroll 1
                    for (int i = 1; i < KK; ++i) {
                        const unsigned li = list[i];
                        if (li < mv) { mv = li; mi = i; }
                    }
                    list[mi] = kk;
                    mv = list[0];
#pragma unroll 1
                    for (int i = 1; i < KK; ++i) {
                        const unsigned li = list[i];
                        if (li < mv) mv = li;
                    }
                    curMin = mv;
                }
            }
        }

        wait_async0();     // prefetched tile landed (per-wave ASYNCcnt)
        __syncthreads();   // all waves' prefetch + selection done
        buf ^= 1;
    }

    // dump this slice's partial top-K key list
    if (tid < QB) {
        unsigned* dst = partial +
            (((long)l * BB + qbase + tid) * NSLICE + slice) * KK;
#pragma unroll 1
        for (int i = 0; i < KK; ++i) dst[i] = s_list[tid * KK + i];
    }
}

// ---------------------------------------------------------------------------
// Merge NSLICE partial top-75 lists per query -> exact top-75 -> class counts.
// One thread per (l, b); 64 threads/block with lists in LDS.
// ---------------------------------------------------------------------------
__global__ __launch_bounds__(64)
void merge_kernel(const unsigned* __restrict__ partial,
                  int* __restrict__ counts /* [L*B][C] */) {
    __shared__ unsigned s_list[64 * KK];
    const int  tid = threadIdx.x;
    const long qid = (long)blockIdx.x * 64 + tid;
    unsigned* list = &s_list[tid * KK];
    const unsigned* src = partial + qid * (NSLICE * KK);

    unsigned curMin = 0xFFFFFFFFu;
#pragma unroll 1
    for (int i = 0; i < KK; ++i) {                 // seed with slice 0's list
        const unsigned k = src[i];
        list[i] = k;
        if (k < curMin) curMin = k;
    }
#pragma unroll 1
    for (int j = KK; j < NSLICE * KK; ++j) {
        const unsigned kk = src[j];
        if (kk > curMin) {
            int mi = 0; unsigned mv = list[0];
#pragma unroll 1
            for (int i = 1; i < KK; ++i) {
                const unsigned li = list[i];
                if (li < mv) { mv = li; mi = i; }
            }
            list[mi] = kk;
            mv = list[0];
#pragma unroll 1
            for (int i = 1; i < KK; ++i) {
                const unsigned li = list[i];
                if (li < mv) mv = li;
            }
            curMin = mv;
        }
    }

    int cnt[CC];
#pragma unroll
    for (int c = 0; c < CC; ++c) cnt[c] = 0;
#pragma unroll 1
    for (int i = 0; i < KK; ++i) {
        const int lab = (int)(list[i] & 15u);
        if (lab < CC) cnt[lab]++;
    }
    int* dst = counts + qid * CC;
#pragma unroll
    for (int c = 0; c < CC; ++c) dst[c] = cnt[c];
}

// ---------------------------------------------------------------------------
// Conformal p-values -> creds. One thread per query.
// ---------------------------------------------------------------------------
__global__ __launch_bounds__(256)
void finalize_kernel(const int* __restrict__ counts, const int* __restrict__ cali,
                     float* __restrict__ out) {
    __shared__ int s_cali[NB_CALI];
    for (int i = threadIdx.x; i < NB_CALI; i += blockDim.x) s_cali[i] = cali[i];
    __syncthreads();

    const int b = blockIdx.x * blockDim.x + threadIdx.x;
    if (b >= BB) return;

    float best = -1.f;
    int   barg = 0;
#pragma unroll
    for (int c = 0; c < CC; ++c) {
        int cnt = 0;
#pragma unroll
        for (int l = 0; l < LL; ++l) cnt += counts[((long)l * BB + b) * CC + c];
        const int nnc = LL * KK - cnt;
        // searchsorted(..., side='left') == lower_bound
        int lo = 0, hi = NB_CALI;
        while (lo < hi) {
            const int mid = (lo + hi) >> 1;
            if (s_cali[mid] < nnc) lo = mid + 1; else hi = mid;
        }
        const float p = (float)(NB_CALI - lo) / (float)NB_CALI;
        if (p > best) { best = p; barg = c; }   // strict '>' => first argmax
    }
#pragma unroll
    for (int c = 0; c < CC; ++c) out[b * CC + c] = (c == barg) ? best : 0.f;
}

// ---------------------------------------------------------------------------
extern "C" void kernel_launch(void* const* d_in, const int* in_sizes, int n_in,
                              void* d_out, int out_size, void* d_ws, size_t ws_size,
                              hipStream_t stream) {
    const float* dact    = (const float*)d_in[0];  // [L,B,D]
    const float* tact    = (const float*)d_in[1];  // [L,N,D]
    const float* centers = (const float*)d_in[2];  // [L,D]
    const int*   labels  = (const int*)d_in[3];    // [N]
    const int*   cali    = (const int*)d_in[4];    // [NB_CALI] sorted
    float*       out     = (float*)d_out;          // [B,C]

    char*  ws  = (char*)d_ws;
    size_t off = 0;
    auto alloc = [&](size_t bytes) -> void* {
        off = (off + 255) & ~(size_t)255;
        void* p = ws + off;
        off += bytes;
        return p;
    };

    half_t*   thi     = (half_t*)  alloc((size_t)LL * NN * DD * sizeof(half_t));
    half_t*   tlo     = (half_t*)  alloc((size_t)LL * NN * DD * sizeof(half_t));
    float*    t2g     = (float*)   alloc((size_t)LL * NN * sizeof(float));
    half_t*   qhi     = (half_t*)  alloc((size_t)LL * BB * DD * sizeof(half_t));
    half_t*   qlo     = (half_t*)  alloc((size_t)LL * BB * DD * sizeof(half_t));
    unsigned* partial = (unsigned*)alloc((size_t)LL * BB * NSLICE * KK * sizeof(unsigned));
    int*      counts  = (int*)     alloc((size_t)LL * BB * CC * sizeof(int));
    (void)ws_size; (void)in_sizes; (void)n_in; (void)out_size;

    {   // train prep: L*N rows, 8 wave32 rows per 256-thread block
        const int rows = LL * NN;
        prep_rows<<<(rows + 7) / 8, 256, 0, stream>>>(
            tact, centers, thi, tlo, t2g, NN, rows);
    }
    {   // query prep
        const int rows = LL * BB;
        prep_rows<<<(rows + 7) / 8, 256, 0, stream>>>(
            dact, centers, qhi, qlo, (float*)nullptr, BB, rows);
    }

    // fused GEMM + selection: 91 KB dynamic LDS (gfx1250: up to 320 KB/WG)
    hipFuncSetAttribute((const void*)knn_kernel,
                        hipFuncAttributeMaxDynamicSharedMemorySize, SMEM_BYTES);
    dim3 grid(BB / QB, LL, NSLICE);   // (16, 4, 4) = 256 blocks, 1024 waves
    knn_kernel<<<grid, WAVES * 32, SMEM_BYTES, stream>>>(
        qhi, qlo, thi, tlo, t2g, labels, partial);

    merge_kernel<<<(LL * BB) / 64, 64, 0, stream>>>(partial, counts);
    finalize_kernel<<<(BB + 255) / 256, 256, 0, stream>>>(counts, cali, out);
}
